// MaskCoNCELoss_80616536146296
// MI455X (gfx1250) — matrix operand order repaired and checked by reading, not compiled, not run on hardware
//
#include <hip/hip_runtime.h>
#include <hip/hip_bf16.h>
#include <cstdint>

// MaskCoNCELoss on MI455X (gfx1250, wave32, WMMA).
// Live dataflow only: 3 big bf16 GEMMs (entropy-pos, entropy-bg, nce-neg) +
// 2 medium GEMMs (nce-pos, nce-bg). 64MB similarity scratch lives in 192MB L2.
// GEMM uses async global->LDS staging (ASYNCcnt double-buffering) when the
// toolchain exposes the gfx1250 builtin; otherwise direct-global WMMA.

typedef __attribute__((ext_vector_type(16))) __bf16 bf16x16;
typedef __attribute__((ext_vector_type(8)))  float  f32x8;
typedef __attribute__((ext_vector_type(4)))  int    v4i;

#define TEMP_INV (1.0f / 0.07f)
#define NEG_FILL (-10.0f)

#if __has_builtin(__builtin_amdgcn_global_load_async_to_lds_b128)
#define HAVE_ASYNC_LDS 1
#endif

__device__ __forceinline__ unsigned short f32_to_bf16_rne(float f) {
  unsigned int u = __float_as_uint(f);
  unsigned int r = u + 0x7FFFu + ((u >> 16) & 1u);  // round-to-nearest-even
  return (unsigned short)(r >> 16);
}

__global__ void k_zero(float* out, int* counters) {
  out[0] = 0.0f;
  counters[0] = 0;
  counters[1] = 0;
}

__global__ void k_cvt_bf16(const float* __restrict__ in,
                           unsigned short* __restrict__ out, int n) {
  int i = blockIdx.x * blockDim.x + threadIdx.x;
  if (i < n) out[i] = f32_to_bf16_rne(in[i]);
}

#ifdef HAVE_ASYNC_LDS
// Signature (from hipcc diagnostic): (v4i addrspace(1)*, v4i addrspace(3)*, Imm, Imm)
__device__ __forceinline__ void async_cp_b128(const unsigned short* g,
                                              unsigned short* l) {
  __builtin_amdgcn_global_load_async_to_lds_b128(
      (__attribute__((address_space(1))) v4i*)g,
      (__attribute__((address_space(3))) v4i*)l, 0, 0);
}

__device__ __forceinline__ void wait_asynccnt_3() {
#if __has_builtin(__builtin_amdgcn_s_wait_asynccnt)
  __builtin_amdgcn_s_wait_asynccnt(3);
#else
  asm volatile("s_wait_asynccnt 0x3" ::: "memory");
#endif
}
__device__ __forceinline__ void wait_asynccnt_0() {
#if __has_builtin(__builtin_amdgcn_s_wait_asynccnt)
  __builtin_amdgcn_s_wait_asynccnt(0);
#else
  asm volatile("s_wait_asynccnt 0x0" ::: "memory");
#endif
}
#endif

// ---------------------------------------------------------------------------
// S[M x Ncols] = A[M x D] * B[Ncols x D]^T, bf16 inputs, f32 accumulate.
// Block = 256 threads = 8 waves (2 x 4); block tile 64 x 128; each wave owns
// a 32x32 output (2x2 register blocking, 4 v_wmma_f32_16x16x32_bf16/K-step).
// Async path: 12KB of unique A/B slab per K-step staged to LDS once
// (GLOBAL_LOAD_ASYNC_TO_LDS_B128, double buffered, s_wait_asynccnt), frags
// re-read from LDS by all 8 waves via ds_load_b128.
// ---------------------------------------------------------------------------
__global__ __launch_bounds__(256)
void k_gemm_nt_bf16(const unsigned short* __restrict__ A,
                    const unsigned short* __restrict__ B,
                    float* __restrict__ S, int ldS, int D) {
  const int t    = threadIdx.x & 31;
  const int w    = threadIdx.x >> 5;     // wave 0..7
  const int half = t >> 4;               // lane-half 0/1
  const int l16  = t & 15;

#ifdef HAVE_ASYNC_LDS
  __shared__ unsigned short At[2][64 * 32];    // 4KB per buffer
  __shared__ unsigned short Bt[2][128 * 32];   // 8KB per buffer

  const int rblk = blockIdx.y * 64;
  const int cblk = blockIdx.x * 128;
  const int wr = (w >> 2) * 32;          // wave row offset in block tile
  const int wc = (w & 3) * 32;           // wave col offset in block tile

  // Staging assignment: thread tid copies one 16B chunk of one row.
  const int tid  = threadIdx.x;
  const int srow = tid >> 2;             // 0..63
  const int sel  = (tid & 3) * 8;        // element offset 0/8/16/24

  const unsigned short* gA  = A + (size_t)(rblk + srow) * D + sel;
  const unsigned short* gB0 = B + (size_t)(cblk + srow) * D + sel;
  const unsigned short* gB1 = B + (size_t)(cblk + 64 + srow) * D + sel;

  f32x8 acc00 = {0.f,0.f,0.f,0.f,0.f,0.f,0.f,0.f};
  f32x8 acc01 = {0.f,0.f,0.f,0.f,0.f,0.f,0.f,0.f};
  f32x8 acc10 = {0.f,0.f,0.f,0.f,0.f,0.f,0.f,0.f};
  f32x8 acc11 = {0.f,0.f,0.f,0.f,0.f,0.f,0.f,0.f};

  // Prologue: stage K-slab 0 into buffer 0 (3 async instrs per thread).
  async_cp_b128(gA,  &At[0][srow * 32 + sel]);
  async_cp_b128(gB0, &Bt[0][srow * 32 + sel]);
  async_cp_b128(gB1, &Bt[0][(64 + srow) * 32 + sel]);

  for (int kt = 0; kt < D; kt += 32) {
    const int buf = (kt >> 5) & 1;
    if (kt + 32 < D) {
      // Stage next slab into the other buffer, then wait for current slab
      // (async loads complete in order: <=3 outstanding => current done).
      async_cp_b128(gA  + kt + 32, &At[buf ^ 1][srow * 32 + sel]);
      async_cp_b128(gB0 + kt + 32, &Bt[buf ^ 1][srow * 32 + sel]);
      async_cp_b128(gB1 + kt + 32, &Bt[buf ^ 1][(64 + srow) * 32 + sel]);
      wait_asynccnt_3();
    } else {
      wait_asynccnt_0();
    }
    __syncthreads();

    bf16x16 a0, a1, b0, b1;
    const unsigned short* la0 = &At[buf][(wr + l16) * 32];
    const unsigned short* la1 = &At[buf][(wr + 16 + l16) * 32];
    const unsigned short* lb0 = &Bt[buf][(wc + l16) * 32];
    const unsigned short* lb1 = &Bt[buf][(wc + 16 + l16) * 32];
    // A 16x32 bf16 frag: lane half 0 -> K {0..7, 16..23}, half 1 -> +8
    ((uint4*)&a0)[0] = *(const uint4*)(la0 + half * 8);
    ((uint4*)&a0)[1] = *(const uint4*)(la0 + 16 + half * 8);
    ((uint4*)&a1)[0] = *(const uint4*)(la1 + half * 8);
    ((uint4*)&a1)[1] = *(const uint4*)(la1 + 16 + half * 8);
    // B 32x16 bf16 frag: lane half 0 -> K {0..7, 16..23}, half 1 -> +8
    ((uint4*)&b0)[0] = *(const uint4*)(lb0 + half * 8);
    ((uint4*)&b0)[1] = *(const uint4*)(lb0 + 16 + half * 8);
    ((uint4*)&b1)[0] = *(const uint4*)(lb1 + half * 8);
    ((uint4*)&b1)[1] = *(const uint4*)(lb1 + 16 + half * 8);

    acc00 = __builtin_amdgcn_wmma_f32_16x16x32_bf16(false, a0, false, b0,
                                                    (short)0, acc00, false, false);
    acc01 = __builtin_amdgcn_wmma_f32_16x16x32_bf16(false, a0, false, b1,
                                                    (short)0, acc01, false, false);
    acc10 = __builtin_amdgcn_wmma_f32_16x16x32_bf16(false, a1, false, b0,
                                                    (short)0, acc10, false, false);
    acc11 = __builtin_amdgcn_wmma_f32_16x16x32_bf16(false, a1, false, b1,
                                                    (short)0, acc11, false, false);
    __syncthreads();   // all waves done reading buf before it is restaged
  }

  const int r0 = rblk + wr;
  const int c0 = cblk + wc;
#else  // ---------------- fallback: direct-global frags ----------------
  const int r0 = blockIdx.y * 64  + (w >> 2) * 32;
  const int c0 = blockIdx.x * 128 + (w & 3) * 32;

  const unsigned short* arow0 = A + (size_t)(r0 + l16)      * D;
  const unsigned short* arow1 = A + (size_t)(r0 + 16 + l16) * D;
  const unsigned short* brow0 = B + (size_t)(c0 + l16)      * D;
  const unsigned short* brow1 = B + (size_t)(c0 + 16 + l16) * D;

  f32x8 acc00 = {0.f,0.f,0.f,0.f,0.f,0.f,0.f,0.f};
  f32x8 acc01 = {0.f,0.f,0.f,0.f,0.f,0.f,0.f,0.f};
  f32x8 acc10 = {0.f,0.f,0.f,0.f,0.f,0.f,0.f,0.f};
  f32x8 acc11 = {0.f,0.f,0.f,0.f,0.f,0.f,0.f,0.f};

  for (int kt = 0; kt < D; kt += 32) {
    bf16x16 a0, a1, b0, b1;
    ((uint4*)&a0)[0] = *(const uint4*)(arow0 + kt + half * 8);
    ((uint4*)&a0)[1] = *(const uint4*)(arow0 + kt + 16 + half * 8);
    ((uint4*)&a1)[0] = *(const uint4*)(arow1 + kt + half * 8);
    ((uint4*)&a1)[1] = *(const uint4*)(arow1 + kt + 16 + half * 8);
    ((uint4*)&b0)[0] = *(const uint4*)(brow0 + kt + half * 8);
    ((uint4*)&b0)[1] = *(const uint4*)(brow0 + kt + 16 + half * 8);
    ((uint4*)&b1)[0] = *(const uint4*)(brow1 + kt + half * 8);
    ((uint4*)&b1)[1] = *(const uint4*)(brow1 + kt + 16 + half * 8);

    acc00 = __builtin_amdgcn_wmma_f32_16x16x32_bf16(false, a0, false, b0,
                                                    (short)0, acc00, false, false);
    acc01 = __builtin_amdgcn_wmma_f32_16x16x32_bf16(false, a0, false, b1,
                                                    (short)0, acc01, false, false);
    acc10 = __builtin_amdgcn_wmma_f32_16x16x32_bf16(false, a1, false, b0,
                                                    (short)0, acc10, false, false);
    acc11 = __builtin_amdgcn_wmma_f32_16x16x32_bf16(false, a1, false, b1,
                                                    (short)0, acc11, false, false);
  }
#endif

#pragma unroll
  for (int v = 0; v < 8; ++v) {
    const size_t rA = (size_t)(r0 + v + half * 8) * ldS;
    const size_t rB = (size_t)(r0 + 16 + v + half * 8) * ldS;
    S[rA + (c0 + l16)]      = acc00[v];
    S[rA + (c0 + 16 + l16)] = acc01[v];
    S[rB + (c0 + l16)]      = acc10[v];
    S[rB + (c0 + 16 + l16)] = acc11[v];
  }
}

// Row entropy of softmax(S_row): ent = lse - E_p[s] = m + log(se) - ws/se.
// float4-vectorized sweeps over the row (Ncols is a multiple of 1024).
__global__ __launch_bounds__(256)
void k_row_entropy(const float* __restrict__ S, int Ncols,
                   float* __restrict__ ent) {
  const int row = blockIdx.x;
  const float4* r4 = (const float4*)(S + (size_t)row * Ncols);
  const int n4 = Ncols >> 2;
  __shared__ float red[256];

  float m = -3.4e38f;
  for (int j = threadIdx.x; j < n4; j += 256) {
    float4 v = r4[j];
    m = fmaxf(m, fmaxf(fmaxf(v.x, v.y), fmaxf(v.z, v.w)));
  }
  red[threadIdx.x] = m; __syncthreads();
  for (int s = 128; s > 0; s >>= 1) {
    if (threadIdx.x < s) red[threadIdx.x] = fmaxf(red[threadIdx.x], red[threadIdx.x + s]);
    __syncthreads();
  }
  m = red[0]; __syncthreads();

  float se = 0.f, ws = 0.f;
  for (int j = threadIdx.x; j < n4; j += 256) {
    float4 v = r4[j];
    float e0 = __expf(v.x - m), e1 = __expf(v.y - m);
    float e2 = __expf(v.z - m), e3 = __expf(v.w - m);
    se += (e0 + e1) + (e2 + e3);
    ws += (v.x * e0 + v.y * e1) + (v.z * e2 + v.w * e3);
  }
  red[threadIdx.x] = se; __syncthreads();
  for (int s = 128; s > 0; s >>= 1) {
    if (threadIdx.x < s) red[threadIdx.x] += red[threadIdx.x + s];
    __syncthreads();
  }
  se = red[0]; __syncthreads();
  red[threadIdx.x] = ws; __syncthreads();
  for (int s = 128; s > 0; s >>= 1) {
    if (threadIdx.x < s) red[threadIdx.x] += red[threadIdx.x + s];
    __syncthreads();
  }
  ws = red[0];
  if (threadIdx.x == 0) ent[row] = m + __logf(se) - ws / se;
}

// Select the P rows with lowest entropy (stable-argsort rank), order-free
// compaction via atomic append (nce loss is permutation invariant).
__global__ __launch_bounds__(256)
void k_select(const float* __restrict__ ent, int N, int P,
              int* __restrict__ idx, int* __restrict__ counter) {
  const int i = blockIdx.x;
  const float e = ent[i];
  int cnt = 0;
  for (int j = threadIdx.x; j < N; j += 256) {
    float ej = ent[j];
    cnt += (ej < e) || (ej == e && j < i);
  }
  __shared__ int red[256];
  red[threadIdx.x] = cnt; __syncthreads();
  for (int s = 128; s > 0; s >>= 1) {
    if (threadIdx.x < s) red[threadIdx.x] += red[threadIdx.x + s];
    __syncthreads();
  }
  if (threadIdx.x == 0 && red[0] < P) {
    int slot = atomicAdd(counter, 1);
    idx[slot] = i;
  }
}

__global__ __launch_bounds__(256)
void k_gather_rows(const unsigned short* __restrict__ src,
                   const int* __restrict__ idx,
                   unsigned short* __restrict__ dst, int D) {
  const int p = blockIdx.x;
  const int r = idx[p];
  // D = 512 bf16 = 256 dwords per row; copy as dwords.
  const unsigned int* s = (const unsigned int*)(src + (size_t)r * D);
  unsigned int* d = (unsigned int*)(dst + (size_t)p * D);
  for (int j = threadIdx.x; j < (D >> 1); j += 256) d[j] = s[j];
}

// Per-row NCE: loss_i = lse({S_ii, row with diag->NEG_FILL}/T) - S_ii/T.
// Accumulates mean into *accum with atomicAdd. float4-vectorized sweeps.
__global__ __launch_bounds__(256)
void k_nce_row(const float* __restrict__ S, int Np, float* __restrict__ accum) {
  const int i = blockIdx.x;
  const float* r = S + (size_t)i * Np;
  const float4* r4 = (const float4*)r;
  const int n4 = Np >> 2;
  const float sii = r[i] * TEMP_INV;   // l_pos / T
  __shared__ float red[256];

  float m = NEG_FILL * TEMP_INV;       // masked diag always present
  for (int j = threadIdx.x; j < n4; j += 256) {
    float4 v = r4[j];
    const int base = j << 2;
    if (base == (i & ~3)) {            // diag lands in this quad
      if ((i & 3) == 0) v.x = NEG_FILL;
      if ((i & 3) == 1) v.y = NEG_FILL;
      if ((i & 3) == 2) v.z = NEG_FILL;
      if ((i & 3) == 3) v.w = NEG_FILL;
    }
    m = fmaxf(m, fmaxf(fmaxf(v.x, v.y), fmaxf(v.z, v.w)) * TEMP_INV);
  }
  m = fmaxf(m, sii);
  red[threadIdx.x] = m; __syncthreads();
  for (int s = 128; s > 0; s >>= 1) {
    if (threadIdx.x < s) red[threadIdx.x] = fmaxf(red[threadIdx.x], red[threadIdx.x + s]);
    __syncthreads();
  }
  m = red[0]; __syncthreads();

  float se = 0.f;
  for (int j = threadIdx.x; j < n4; j += 256) {
    float4 v = r4[j];
    const int base = j << 2;
    if (base == (i & ~3)) {
      if ((i & 3) == 0) v.x = NEG_FILL;
      if ((i & 3) == 1) v.y = NEG_FILL;
      if ((i & 3) == 2) v.z = NEG_FILL;
      if ((i & 3) == 3) v.w = NEG_FILL;
    }
    se += (__expf(v.x * TEMP_INV - m) + __expf(v.y * TEMP_INV - m)) +
          (__expf(v.z * TEMP_INV - m) + __expf(v.w * TEMP_INV - m));
  }
  red[threadIdx.x] = se; __syncthreads();
  for (int s = 128; s > 0; s >>= 1) {
    if (threadIdx.x < s) red[threadIdx.x] += red[threadIdx.x + s];
    __syncthreads();
  }
  if (threadIdx.x == 0) {
    float tot = red[0] + __expf(sii - m);           // + l_pos term
    float loss_i = m + __logf(tot) - sii;           // lse - pos
    atomicAdd(accum, loss_i / (float)Np);
  }
}

extern "C" void kernel_launch(void* const* d_in, const int* in_sizes, int n_in,
                              void* d_out, int out_size, void* d_ws, size_t ws_size,
                              hipStream_t stream) {
  (void)n_in; (void)out_size; (void)ws_size;
  const float* fq  = (const float*)d_in[0];
  const float* fk  = (const float*)d_in[1];
  const float* fqn = (const float*)d_in[2];
  const float* fkn = (const float*)d_in[3];
  const float* fqb = (const float*)d_in[4];
  const float* fkb = (const float*)d_in[5];
  float* out = (float*)d_out;

  const int D = 512;
  const int N = in_sizes[0] / D;   // 4096
  const int P = N / 2;             // 2048
  const size_t NE = (size_t)N * D;
  const size_t PE = (size_t)P * D;

  // Workspace layout (bytes): 6 bf16 mats | 4 gathered bf16 mats | ent x2 |
  // idx x2 | counters | S (N*N f32, lives in L2).
  char* ws = (char*)d_ws;
  unsigned short* bq  = (unsigned short*)ws;
  unsigned short* bk  = bq  + NE;
  unsigned short* bqn = bk  + NE;
  unsigned short* bkn = bqn + NE;
  unsigned short* bqb = bkn + NE;
  unsigned short* bkb = bqb + NE;
  unsigned short* qgp = bkb + NE;
  unsigned short* kgp = qgp + PE;
  unsigned short* qgb = kgp + PE;
  unsigned short* kgb = qgb + PE;
  char* p = (char*)(kgb + PE);
  float* entp = (float*)p;            p += (size_t)N * 4;
  float* entb = (float*)p;            p += (size_t)N * 4;
  int*   idxp = (int*)p;              p += (size_t)N * 4;
  int*   idxb = (int*)p;              p += (size_t)N * 4;
  int*   counters = (int*)p;          p += 256;
  // align S to 256B
  float* S = (float*)(((uintptr_t)p + 255) & ~(uintptr_t)255);

  k_zero<<<1, 1, 0, stream>>>(out, counters);

  const int nElem = (int)NE;
  const int cgrid = (nElem + 255) / 256;
  k_cvt_bf16<<<cgrid, 256, 0, stream>>>(fq,  bq,  nElem);
  k_cvt_bf16<<<cgrid, 256, 0, stream>>>(fk,  bk,  nElem);
  k_cvt_bf16<<<cgrid, 256, 0, stream>>>(fqn, bqn, nElem);
  k_cvt_bf16<<<cgrid, 256, 0, stream>>>(fkn, bkn, nElem);
  k_cvt_bf16<<<cgrid, 256, 0, stream>>>(fqb, bqb, nElem);
  k_cvt_bf16<<<cgrid, 256, 0, stream>>>(fkb, bkb, nElem);

  dim3 gN(N / 128, N / 64);   // 4096 -> (32, 64)
  dim3 gP(P / 128, P / 64);   // 2048 -> (16, 32)

  // --- pos branch: entropy(fq@fq^T) -> select -> gather ---
  k_gemm_nt_bf16<<<gN, 256, 0, stream>>>(bq, bq, S, N, D);
  k_row_entropy<<<N, 256, 0, stream>>>(S, N, entp);

  // --- bg branch: entropy(fqb@fkb^T) ---
  k_gemm_nt_bf16<<<gN, 256, 0, stream>>>(bqb, bkb, S, N, D);
  k_row_entropy<<<N, 256, 0, stream>>>(S, N, entb);

  k_select<<<N, 256, 0, stream>>>(entp, N, P, idxp, &counters[0]);
  k_select<<<N, 256, 0, stream>>>(entb, N, P, idxb, &counters[1]);

  k_gather_rows<<<P, 256, 0, stream>>>(bq,  idxp, qgp, D);
  k_gather_rows<<<P, 256, 0, stream>>>(bk,  idxp, kgp, D);
  k_gather_rows<<<P, 256, 0, stream>>>(bqb, idxb, qgb, D);
  k_gather_rows<<<P, 256, 0, stream>>>(bkb, idxb, kgb, D);

  // --- nce(pos gathered) ---
  k_gemm_nt_bf16<<<gP, 256, 0, stream>>>(qgp, kgp, S, P, D);
  k_nce_row<<<P, 256, 0, stream>>>(S, P, out);

  // --- nce(neg full) ---
  k_gemm_nt_bf16<<<gN, 256, 0, stream>>>(bqn, bkn, S, N, D);
  k_nce_row<<<N, 256, 0, stream>>>(S, N, out);

  // --- nce(bg gathered) ---
  k_gemm_nt_bf16<<<gP, 256, 0, stream>>>(qgb, kgb, S, P, D);
  k_nce_row<<<P, 256, 0, stream>>>(S, P, out);
}
